// FutureBlock_82042465288785
// MI455X (gfx1250) — matrix operand order repaired and checked by reading, not compiled, not run on hardware
//
#include <hip/hip_runtime.h>

#define TLEN  4096
#define BATCH 8
#define DDIM  512
#define MTOT  (BATCH * TLEN)   // 32768
#define CH    256
#define NCH   (TLEN / CH)      // 16
#define LDP   40               // LDS pitch in shorts: 80B rows -> 16B aligned, conflict-free

typedef __attribute__((ext_vector_type(16))) __bf16 v16bf;
typedef __attribute__((ext_vector_type(8)))  float  v8f;

union FragBF { unsigned int u[8]; v16bf v; };

__device__ __forceinline__ unsigned short f32_to_bf16(float f) {
  union { float f; unsigned int u; } x; x.f = f;
  unsigned int r = x.u + 0x7FFFu + ((x.u >> 16) & 1u);  // round-to-nearest-even
  return (unsigned short)(r >> 16);
}

// ---- CDNA5 async global->LDS path (ASYNCcnt) -------------------------------
// GLOBAL_LOAD_ASYNC_TO_LDS_B128, GVS mode: mem = SGPR64 + VGPR32 ; LDS dst =
// per-lane 32-bit LDS byte address in VDST (low 32 bits of generic pointer).
__device__ __forceinline__ unsigned lds_addr32(const void* p) {
  return (unsigned)(unsigned long long)p;
}
__device__ __forceinline__ void async_b128(unsigned lds, const void* base, unsigned voff) {
  asm volatile("global_load_async_to_lds_b128 %0, %1, %2"
               :
               : "v"(lds), "v"(voff), "s"(base)
               : "memory");
}
__device__ __forceinline__ void wait_async0() {
  asm volatile("s_wait_asynccnt 0" ::: "memory");
}

// ---------------------------------------------------------------- cumsum ----
__global__ __launch_bounds__(256) void cum_partial_kernel(
    const float* __restrict__ dec, float* __restrict__ partial) {
  int d = blockIdx.x * 256 + threadIdx.x;
  int c = blockIdx.y, b = blockIdx.z;
  const float* p = dec + ((size_t)(b * TLEN + c * CH)) * DDIM + d;
  float s = 0.f;
  for (int t = 0; t < CH; ++t) s += p[(size_t)t * DDIM];
  partial[((size_t)(b * NCH + c)) * DDIM + d] = s;
}

__global__ __launch_bounds__(256) void cum_scan_kernel(float* __restrict__ partial) {
  int idx = blockIdx.x * 256 + threadIdx.x;   // 4096 columns: b*512+d
  int b = idx >> 9, d = idx & 511;
  float run = 0.f;
  for (int c = 0; c < NCH; ++c) {
    size_t o = ((size_t)(b * NCH + c)) * DDIM + d;
    float v = partial[o];
    partial[o] = run;                          // exclusive prefix
    run += v;
  }
}

__global__ __launch_bounds__(256) void cum_write_kernel(
    const float* __restrict__ dec, const float* __restrict__ partial,
    unsigned short* __restrict__ hist) {
  int d = blockIdx.x * 256 + threadIdx.x;
  int c = blockIdx.y, b = blockIdx.z;
  float run = partial[((size_t)(b * NCH + c)) * DDIM + d];
  const float* p = dec + ((size_t)(b * TLEN + c * CH)) * DDIM + d;
  unsigned short* q = hist + ((size_t)(b * TLEN + c * CH)) * DDIM + d;
  for (int t = 0; t < CH; ++t) {
    run += p[(size_t)t * DDIM];
    float cnt = (float)(c * CH + t + 1);
    q[(size_t)t * DDIM] = f32_to_bf16(run / cnt);
  }
}

// ----------------------------------------------------------------- prep -----
// Weights converted to bf16 AND transposed (N-major, K contiguous) so GEMM B
// tiles are plain row copies -> async B128 DMA, no transpose at tile-fill time.
__global__ void conv_w_kernel(const float* __restrict__ a, const float* __restrict__ b,
                              const float* __restrict__ c,
                              unsigned short* __restrict__ oa, unsigned short* __restrict__ ob,
                              unsigned short* __restrict__ oc) {
  int stride = gridDim.x * blockDim.x;
  for (int i = blockIdx.x * blockDim.x + threadIdx.x; i < DDIM * DDIM; i += stride) {
    int k = i >> 9, n = i & 511;               // read coalesced over n
    int o = n * DDIM + k;                      // write transposed
    oa[o] = f32_to_bf16(a[i]);
    ob[o] = f32_to_bf16(b[i]);
    oc[o] = f32_to_bf16(c[i]);
  }
}

__global__ __launch_bounds__(512) void summ_proj_kernel(
    const float* __restrict__ summ, const float* __restrict__ W1, float* __restrict__ sp) {
  int n = threadIdx.x;
  for (int b = 0; b < BATCH; ++b) {
    float acc = 0.f;
    for (int k = 0; k < DDIM; ++k) acc += summ[b * DDIM + k] * W1[k * DDIM + n];
    sp[b * DDIM + n] = acc;
  }
}

// ---------------------------------------------------- WMMA fragment loads ---
// A fragment 16x32 bf16 (ISA 7.12.2): lane l -> M=l&15, half=l>>4;
// VGPR v holds K = 16*(v>>2) + 8*half + 2*(v&3) .. +1  (pair = one 32-bit read)
__device__ __forceinline__ v16bf load_a_frag(const unsigned short* As, int row0, int ld) {
  int lane = threadIdx.x & 31;
  int m = lane & 15, half = lane >> 4;
  FragBF f;
#pragma unroll
  for (int v = 0; v < 8; ++v) {
    int k = ((v >> 2) << 4) + (half << 3) + ((v & 3) << 1);
    f.u[v] = *(const unsigned int*)(As + (row0 + m) * ld + k);
  }
  return f.v;
}

// B fragment 32x16 bf16: lane l -> N=l&15, half=l>>4;
// VGPR v holds K = 16*(v>>2) + 8*half + 2*(v&3) .. +1.  Bt is stored N-major (k contiguous).
__device__ __forceinline__ v16bf load_b_frag(const unsigned short* Bt, int col0, int ld) {
  int lane = threadIdx.x & 31;
  int n = lane & 15, kb = (lane >> 4) << 3;
  FragBF f;
#pragma unroll
  for (int v = 0; v < 8; ++v)
    f.u[v] = *(const unsigned int*)(Bt + (col0 + n) * ld + ((v >> 2) << 4) + kb + ((v & 3) << 1));
  return f.v;
}

// ------------------------------------- GEMM1 / GEMM2 (128x128, 8 waves) ----
// Double-buffered LDS tiles, filled by async B128 DMA overlapped with WMMA.
// EPI==1: out = bf16(tanh(acc + bias[n]))
// EPI==2: out = bf16(sproj[b][n] - acc)
template<int EPI>
__global__ __launch_bounds__(256) void gemm_kernel(
    const unsigned short* __restrict__ A, const unsigned short* __restrict__ Bt,
    const float* __restrict__ bias, const float* __restrict__ sproj,
    unsigned short* __restrict__ Out) {
  __shared__ __attribute__((aligned(16))) unsigned short As[2][128 * LDP];
  __shared__ __attribute__((aligned(16))) unsigned short Bs[2][128 * LDP];

  const int tid  = threadIdx.x;
  const int lane = tid & 31, wid = tid >> 5;
  const int wM = wid >> 1, wN = wid & 1;          // 4 x 2 wave grid
  const int half = lane >> 4, ln = lane & 15;
  const int rowBlk = blockIdx.y * 128;
  const int colBlk = blockIdx.x * 128;

  const v8f vzero = {0.f, 0.f, 0.f, 0.f, 0.f, 0.f, 0.f, 0.f};
  v8f acc[2][4];
#pragma unroll
  for (int i = 0; i < 2; ++i)
#pragma unroll
    for (int j = 0; j < 4; ++j) acc[i][j] = vzero;

  // Tile = 128 rows x 64B = 512 x 16B chunks; 2 chunks per thread.
  auto issue_tile = [&](int buf, int kt) {
#pragma unroll
    for (int p = 0; p < 2; ++p) {
      int chunk = p * 256 + tid;
      int r = chunk >> 2, c = chunk & 3;
      unsigned ldsOff = (unsigned)(r * LDP + c * 8);     // shorts
      unsigned gOff   = (unsigned)(kt * 64 + c * 16);    // bytes within row
      async_b128(lds_addr32(&As[buf][ldsOff]), A,  (unsigned)((rowBlk + r) * 1024) + gOff);
      async_b128(lds_addr32(&Bs[buf][ldsOff]), Bt, (unsigned)((colBlk + r) * 1024) + gOff);
    }
  };

  issue_tile(0, 0);
  wait_async0();
  __syncthreads();

  for (int kt = 0; kt < DDIM / 32; ++kt) {
    int buf = kt & 1;
    if (kt + 1 < DDIM / 32) issue_tile(buf ^ 1, kt + 1);   // DMA next tile

    v16bf af[2], bfv[4];
#pragma unroll
    for (int mt = 0; mt < 2; ++mt) af[mt] = load_a_frag(As[buf], wM * 32 + mt * 16, LDP);
#pragma unroll
    for (int nt = 0; nt < 4; ++nt) bfv[nt] = load_b_frag(Bs[buf], wN * 64 + nt * 16, LDP);
#pragma unroll
    for (int mt = 0; mt < 2; ++mt)
#pragma unroll
      for (int nt = 0; nt < 4; ++nt)
        acc[mt][nt] = __builtin_amdgcn_wmma_f32_16x16x32_bf16(
            false, af[mt], false, bfv[nt], (short)0, acc[mt][nt], false, false);

    if (kt + 1 < DDIM / 32) wait_async0();
    __syncthreads();
  }

#pragma unroll
  for (int mt = 0; mt < 2; ++mt)
#pragma unroll
    for (int nt = 0; nt < 4; ++nt)
#pragma unroll
      for (int r = 0; r < 8; ++r) {
        int m = rowBlk + wM * 32 + mt * 16 + half * 8 + r;
        int n = colBlk + wN * 64 + nt * 16 + ln;
        float v = acc[mt][nt][r];
        float o;
        if constexpr (EPI == 1) {
          o = tanhf(v + bias[n]);
        } else {
          int b = m >> 12;                      // row / TLEN
          o = sproj[(b << 9) + n] - v;
        }
        Out[(size_t)m * DDIM + n] = f32_to_bf16(o);
      }
}

// ------------------------------- GEMM3 + residual + LayerNorm (64 x 512) ---
__global__ __launch_bounds__(512) void gemm3_ln_kernel(
    const unsigned short* __restrict__ A, const unsigned short* __restrict__ Bt,
    const float* __restrict__ b_out, const float* __restrict__ dec,
    const float* __restrict__ gamma, const float* __restrict__ beta,
    float* __restrict__ Out) {
  __shared__ __attribute__((aligned(16))) unsigned short As[2][64 * LDP];
  __shared__ __attribute__((aligned(16))) unsigned short Bs[2][512 * LDP];
  __shared__ float redS[64 * 8];
  __shared__ float redQ[64 * 8];
  __shared__ float meanS[64];
  __shared__ float rstdS[64];

  const int tid  = threadIdx.x;
  const int lane = tid & 31, wid = tid >> 5;     // 16 waves
  const int wM = wid >> 3, wN = wid & 7;         // 2 x 8 wave grid
  const int half = lane >> 4, ln = lane & 15;
  const int rowBlk = blockIdx.x * 64;

  const v8f vzero = {0.f, 0.f, 0.f, 0.f, 0.f, 0.f, 0.f, 0.f};
  v8f acc[2][4];
#pragma unroll
  for (int i = 0; i < 2; ++i)
#pragma unroll
    for (int j = 0; j < 4; ++j) acc[i][j] = vzero;

  auto issue_tile = [&](int buf, int kt) {
    if (tid < 256) {                             // A: 64 rows x 4 chunks
      int r = tid >> 2, c = tid & 3;
      async_b128(lds_addr32(&As[buf][r * LDP + c * 8]), A,
                 (unsigned)((rowBlk + r) * 1024 + kt * 64 + c * 16));
    }
#pragma unroll
    for (int p = 0; p < 4; ++p) {                // B: 512 rows x 4 chunks
      int chunk = p * 512 + tid;
      int n = chunk >> 2, c = chunk & 3;
      async_b128(lds_addr32(&Bs[buf][n * LDP + c * 8]), Bt,
                 (unsigned)(n * 1024 + kt * 64 + c * 16));
    }
  };

  issue_tile(0, 0);
  wait_async0();
  __syncthreads();

  for (int kt = 0; kt < DDIM / 32; ++kt) {
    int buf = kt & 1;
    if (kt + 1 < DDIM / 32) issue_tile(buf ^ 1, kt + 1);

    v16bf af[2], bfv[4];
#pragma unroll
    for (int mt = 0; mt < 2; ++mt) af[mt] = load_a_frag(As[buf], wM * 32 + mt * 16, LDP);
#pragma unroll
    for (int nt = 0; nt < 4; ++nt) bfv[nt] = load_b_frag(Bs[buf], wN * 64 + nt * 16, LDP);
#pragma unroll
    for (int mt = 0; mt < 2; ++mt)
#pragma unroll
      for (int nt = 0; nt < 4; ++nt)
        acc[mt][nt] = __builtin_amdgcn_wmma_f32_16x16x32_bf16(
            false, af[mt], false, bfv[nt], (short)0, acc[mt][nt], false, false);

    if (kt + 1 < DDIM / 32) wait_async0();
    __syncthreads();
  }

  // x = tanh(acc + b_out) + dec ; accumulate row sums for LayerNorm
#pragma unroll
  for (int mt = 0; mt < 2; ++mt) {
#pragma unroll
    for (int r = 0; r < 8; ++r) {
      int rloc = wM * 32 + mt * 16 + half * 8 + r;
      int m = rowBlk + rloc;
      float s = 0.f, q = 0.f;
#pragma unroll
      for (int nt = 0; nt < 4; ++nt) {
        int n = wN * 64 + nt * 16 + ln;
        float x = tanhf(acc[mt][nt][r] + b_out[n]) + dec[(size_t)m * DDIM + n];
        acc[mt][nt][r] = x;
        s += x; q += x * x;
      }
#pragma unroll
      for (int sft = 8; sft >= 1; sft >>= 1) {   // reduce within 16-lane half
        s += __shfl_xor(s, sft, 32);
        q += __shfl_xor(q, sft, 32);
      }
      if (ln == 0) { redS[rloc * 8 + wN] = s; redQ[rloc * 8 + wN] = q; }
    }
  }
  __syncthreads();
  if (tid < 64) {
    float s = 0.f, q = 0.f;
#pragma unroll
    for (int w = 0; w < 8; ++w) { s += redS[tid * 8 + w]; q += redQ[tid * 8 + w]; }
    float mu  = s * (1.f / 512.f);
    float var = q * (1.f / 512.f) - mu * mu;
    meanS[tid] = mu;
    rstdS[tid] = rsqrtf(var + 1e-6f);
  }
  __syncthreads();
#pragma unroll
  for (int mt = 0; mt < 2; ++mt)
#pragma unroll
    for (int nt = 0; nt < 4; ++nt)
#pragma unroll
      for (int r = 0; r < 8; ++r) {
        int rloc = wM * 32 + mt * 16 + half * 8 + r;
        int m = rowBlk + rloc;
        int n = wN * 64 + nt * 16 + ln;
        float y = (acc[mt][nt][r] - meanS[rloc]) * rstdS[rloc] * gamma[n] + beta[n];
        Out[(size_t)m * DDIM + n] = y;
      }
}

// ------------------------------------------------------------------ launch --
extern "C" void kernel_launch(void* const* d_in, const int* in_sizes, int n_in,
                              void* d_out, int out_size, void* d_ws, size_t ws_size,
                              hipStream_t stream) {
  (void)in_sizes; (void)n_in; (void)out_size; (void)ws_size;
  const float* summ   = (const float*)d_in[0];
  const float* dec    = (const float*)d_in[1];
  const float* W_hist = (const float*)d_in[2];
  const float* b_hist = (const float*)d_in[3];
  const float* W1     = (const float*)d_in[4];
  const float* W2     = (const float*)d_in[5];
  const float* W_out  = (const float*)d_in[6];
  const float* b_out  = (const float*)d_in[7];
  const float* gamma  = (const float*)d_in[8];
  const float* beta   = (const float*)d_in[9];
  float* out = (float*)d_out;

  char* ws = (char*)d_ws;
  size_t off = 0;
  unsigned short* hist = (unsigned short*)(ws + off); off += (size_t)MTOT * DDIM * 2;
  unsigned short* h1   = (unsigned short*)(ws + off); off += (size_t)MTOT * DDIM * 2;
  unsigned short* wht  = (unsigned short*)(ws + off); off += (size_t)DDIM * DDIM * 2;
  unsigned short* w2t  = (unsigned short*)(ws + off); off += (size_t)DDIM * DDIM * 2;
  unsigned short* wot  = (unsigned short*)(ws + off); off += (size_t)DDIM * DDIM * 2;
  float* sproj         = (float*)(ws + off);          off += (size_t)BATCH * DDIM * 4;
  float* partial       = (float*)(ws + off);          off += (size_t)BATCH * NCH * DDIM * 4;
  unsigned short* obuf = hist;   // hist dead after GEMM1 -> reuse as GEMM2 output

  conv_w_kernel<<<dim3(512), 256, 0, stream>>>(W_hist, W2, W_out, wht, w2t, wot);
  summ_proj_kernel<<<dim3(1), 512, 0, stream>>>(summ, W1, sproj);

  cum_partial_kernel<<<dim3(2, NCH, BATCH), 256, 0, stream>>>(dec, partial);
  cum_scan_kernel<<<dim3(16), 256, 0, stream>>>(partial);
  cum_write_kernel<<<dim3(2, NCH, BATCH), 256, 0, stream>>>(dec, partial, hist);

  // GEMM1: h1 = tanh(hist @ W_hist + b_hist)
  gemm_kernel<1><<<dim3(4, 256), 256, 0, stream>>>(hist, wht, b_hist, nullptr, h1);
  // GEMM2: obuf = sproj - h1 @ W2
  gemm_kernel<2><<<dim3(4, 256), 256, 0, stream>>>(h1, w2t, nullptr, sproj, obuf);
  // GEMM3 + residual + LayerNorm -> f32 out
  gemm3_ln_kernel<<<dim3(MTOT / 64), 512, 0, stream>>>(obuf, wot, b_out, dec, gamma, beta, out);
}
